// MFEM_70952859730211
// MI455X (gfx1250) — compile-verified, matrix operand order checked by reading, not run on hardware
//
#include <hip/hip_runtime.h>

typedef _Float16 h16;
typedef __attribute__((ext_vector_type(16))) _Float16 v16h;
typedef __attribute__((ext_vector_type(8)))  _Float16 v8h;
typedef __attribute__((ext_vector_type(8)))  float    v8f;

#define NPTS  4096
#define BATCH 4
#define KNN   32

struct alignas(16) B16 { unsigned int w[4]; };

__device__ __forceinline__ v8f wmma_f16(v16h a, v16h b, v8f c) {
  return __builtin_amdgcn_wmma_f32_16x16x32_f16(false, a, false, b, (short)0, c, false, false);
}

// 16x32 f16 fragment load (used for both A row-major [M][K] and B column-major
// [N][K] since both are "16 rows x 32 K-halves at stride ldK").
// Lane l holds row/col (l%16); half = l/16 selects the K interleave:
//   element e <-> K = kbase + (e<8 ? e : e+8) + 8*half
// Two aligned 16-byte loads per lane.
__device__ __forceinline__ v16h load_frag(const h16* __restrict__ X, int ldK,
                                          int rbase, int kbase, int lane) {
  const int hf = lane >> 4;
  const int r  = lane & 15;
  const h16* p = X + (size_t)(rbase + r) * ldK + kbase + 8 * hf;
  union { B16 u[2]; v16h v; } t;
  t.u[0] = *reinterpret_cast<const B16*>(p);
  t.u[1] = *reinterpret_cast<const B16*>(p + 16);
  return t.v;
}

// D tile: lane l holds (M = mbase + r + 8*(l/16), N = colbase + l%16) for r=0..7.
// Column-major f16 store: 8 consecutive halves -> one 16-byte store per lane.
__device__ __forceinline__ void store_d_relu_cm(h16* __restrict__ X, int ldK,
                                                int colbase, int mbase, int lane, v8f acc) {
  const int hf  = lane >> 4;
  const int col = lane & 15;
  union { v8h v; B16 u; } pk;
#pragma unroll
  for (int r = 0; r < 8; ++r) pk.v[r] = (h16)fmaxf(acc[r], 0.f);
  *reinterpret_cast<B16*>(X + (size_t)(colbase + col) * ldK + mbase + 8 * hf) = pk.u;
}

// ---------------------------------------------------------------------------
// Prep: convert f32 weight matrices to f16 once (weights are L2-resident and
// reused B*256 times per layer; this removes per-wave f32 loads + v_cvt).
// ---------------------------------------------------------------------------
__global__ __launch_bounds__(256) void cvt_f32_f16_kernel(const float* __restrict__ in,
                                                          h16* __restrict__ out, int n) {
  const int i = blockIdx.x * 256 + threadIdx.x;
  if (i < n) out[i] = (h16)in[i];
}

// ---------------------------------------------------------------------------
// Stage A: exact KNN (top-33 smallest d = |q|^2 + |p|^2 - 2 q.p, stable ties,
// slot 0 is self and is dropped, matching top_k(-d, k+1)[..., 1:]).
// ---------------------------------------------------------------------------
__global__ __launch_bounds__(256) void knn_kernel(const float* __restrict__ x,
                                                  int* __restrict__ idx) {
  __shared__ float sx[256], sy[256], sz[256], ss[256];
  const int b = blockIdx.x >> 4;                 // 16 blocks per batch
  const int n = (blockIdx.x & 15) * 256 + threadIdx.x;
  const float* xb = x + (size_t)b * 3 * NPTS;
  const float qx = xb[n], qy = xb[NPTS + n], qz = xb[2 * NPTS + n];
  const float qs = qx * qx + qy * qy + qz * qz;

  float bd[KNN + 1];
  int   bid[KNN + 1];
#pragma unroll
  for (int i = 0; i <= KNN; ++i) { bd[i] = 3.0e38f; bid[i] = 0; }

  for (int base = 0; base < NPTS; base += 256) {
    __syncthreads();
    const int t = threadIdx.x;
    const float px = xb[base + t];
    const float py = xb[NPTS + base + t];
    const float pz = xb[2 * NPTS + base + t];
    sx[t] = px; sy[t] = py; sz[t] = pz;
    ss[t] = px * px + py * py + pz * pz;
    __syncthreads();
    for (int jj = 0; jj < 256; ++jj) {
      const float d = qs + ss[jj] - 2.f * (qx * sx[jj] + qy * sy[jj] + qz * sz[jj]);
      if (d < bd[KNN]) {
        int pos = KNN;
        while (pos > 0 && bd[pos - 1] > d) {
          bd[pos] = bd[pos - 1]; bid[pos] = bid[pos - 1]; --pos;
        }
        bd[pos] = d; bid[pos] = base + jj;
      }
    }
  }
  int* op = idx + ((size_t)b * NPTS + n) * KNN;
#pragma unroll
  for (int k = 0; k < KNN; ++k) op[k] = bid[k + 1];   // drop self (slot 0)
}

// ---------------------------------------------------------------------------
// Stage B: fused gather + per-scale MLP (3->64->64->128) + max over K=32.
// One workgroup (8 waves) = one (batch, scale, 8-point tile) -> 256 columns.
// Layer0 in VALU (K=3); layers 1/2 as LDS-resident WMMA GEMMs with
// pre-converted f16 weights; wave w owns point w's two 16-col tiles, so the
// neighbor-max is a register max + 4 __shfl_xor steps (relu/max commute).
// ---------------------------------------------------------------------------
__global__ __launch_bounds__(256) void msg_mlp_kernel(
    const float* __restrict__ x, const int* __restrict__ idx,
    const float* __restrict__ sW0, const float* __restrict__ sb0,
    const h16* __restrict__ whsW1, const float* __restrict__ sb1,
    const h16* __restrict__ whsW2, const float* __restrict__ sb2,
    h16* __restrict__ pointbuf) {
  __shared__ __align__(16) h16 H0[256 * 64];   // 32 KB, col-major [col][64]
  __shared__ __align__(16) h16 H1[256 * 64];   // 32 KB

  const int bi   = blockIdx.x;
  const int tile = bi & 511;                   // 512 tiles of 8 points
  const int s    = (bi >> 9) % 3;
  const int b    = bi / (512 * 3);
  const int tid  = threadIdx.x;

  // ---- layer0 + gather: thread t handles column t (point p=t/32, nbr k=t%32)
  {
    const int col = tid;
    const int p = col >> 5, kk = col & 31;
    const int n = tile * 8 + p;
    const int j = idx[((size_t)b * NPTS + n) * KNN + kk];
    const float gx = x[((size_t)b * 3 + 0) * NPTS + j];
    const float gy = x[((size_t)b * 3 + 1) * NPTS + j];
    const float gz = x[((size_t)b * 3 + 2) * NPTS + j];
    const float* w0 = sW0 + s * 64 * 3;        // uniform -> scalar loads
    const float* b0 = sb0 + s * 64;
    h16* h0c = &H0[col * 64];
#pragma unroll 8
    for (int o = 0; o < 64; ++o) {
      const float v = w0[o * 3 + 0] * gx + w0[o * 3 + 1] * gy + w0[o * 3 + 2] * gz + b0[o];
      h0c[o] = (h16)fmaxf(v, 0.f);
    }
  }
  __syncthreads();

  const int wv = tid >> 5, lane = tid & 31, hf = lane >> 4;

  // ---- layer1: H1[64][256] = relu(W1[64][64] * H0 + b1), WMMA
  {
    const h16* W1h = whsW1 + s * 64 * 64;
    const float* b1 = sb1 + s * 64;
    for (int t = 0; t < 2; ++t) {
      const int colbase = (wv * 2 + t) * 16;
#pragma unroll
      for (int mt = 0; mt < 4; ++mt) {
        v8f acc;
#pragma unroll
        for (int r = 0; r < 8; ++r) acc[r] = b1[mt * 16 + 8 * hf + r];
#pragma unroll
        for (int kt = 0; kt < 64; kt += 32) {
          v16h a  = load_frag(W1h, 64, mt * 16, kt, lane);
          v16h bf = load_frag(H0, 64, colbase, kt, lane);
          acc = wmma_f16(a, bf, acc);
        }
        store_d_relu_cm(H1, 64, colbase, mt * 16, lane, acc);
      }
    }
  }
  __syncthreads();

  // ---- layer2: relu(W2[128][64] * H1 + b2), then max over the point's 32 nbrs
  {
    const h16* W2h = whsW2 + s * 128 * 64;
    const float* b2 = sb2 + s * 128;
    const int nglob = tile * 8 + wv;           // wave wv owns point wv
    h16* outcol = pointbuf + ((size_t)b * NPTS + nglob) * 384 + s * 128;
#pragma unroll
    for (int mt = 0; mt < 8; ++mt) {
      v8f a0, a1;
#pragma unroll
      for (int r = 0; r < 8; ++r) { const float bv = b2[mt * 16 + 8 * hf + r]; a0[r] = bv; a1[r] = bv; }
#pragma unroll
      for (int kt = 0; kt < 64; kt += 32) {
        v16h af  = load_frag(W2h, 64, mt * 16, kt, lane);
        v16h bf0 = load_frag(H1, 64, (2 * wv) * 16, kt, lane);
        v16h bf1 = load_frag(H1, 64, (2 * wv + 1) * 16, kt, lane);
        a0 = wmma_f16(af, bf0, a0);
        a1 = wmma_f16(af, bf1, a1);
      }
      float vr[8];
#pragma unroll
      for (int r = 0; r < 8; ++r) {
        float v = fmaxf(fmaxf(a0[r], a1[r]), 0.f);   // relu(max) == max(relu)
#pragma unroll
        for (int m = 1; m < 16; m <<= 1)             // reduce over N within 16-lane half
          v = fmaxf(v, __shfl_xor(v, m, 32));
        vr[r] = v;
      }
      if ((lane & 15) == 0) {                        // lane 0 -> M rows 0..7, lane 16 -> 8..15
        union { v8h v; B16 u; } pk;
#pragma unroll
        for (int r = 0; r < 8; ++r) pk.v[r] = (h16)vr[r];
        *reinterpret_cast<B16*>(outcol + mt * 16 + 8 * hf) = pk.u;
      }
    }
  }
}

// ---------------------------------------------------------------------------
// Generic WMMA GEMM: Y[b] = relu(W[M][K] * X[b] + bias), f16 weights + f16
// column-major activations X [b][4096][K]. One wave per 16(M) x 64(N) tile
// group: the A-fragment is loaded once per k-step and reused across 4 WMMAs
// (4 accumulators = 32 VGPRs), quadrupling wmma per fragment load.
// F32OUT=false -> f16 column-major [b][4096][M]; true -> f32 row-major [b][M][4096].
// ---------------------------------------------------------------------------
#define NBLK 4
template <bool F32OUT>
__global__ __launch_bounds__(256) void gemm_relu(const h16* __restrict__ Wh,
                                                 const float* __restrict__ bias,
                                                 const h16* __restrict__ X,
                                                 void* __restrict__ outp,
                                                 int M, int K, int tiles_m) {
  const int ngrp  = NPTS / (16 * NBLK);        // 64 N-tile groups per batch
  const int total = BATCH * tiles_m * ngrp;
  const int wid = blockIdx.x * 8 + (threadIdx.x >> 5);
  if (wid >= total) return;
  const int lane = threadIdx.x & 31, hf = lane >> 4;
  const int b   = wid / (tiles_m * ngrp);
  const int rem = wid % (tiles_m * ngrp);
  const int mt  = rem / ngrp;
  const int ng  = rem % ngrp;
  const int mbase = mt * 16, colbase = ng * 16 * NBLK;
  const h16* Xb = X + (size_t)b * NPTS * K;

  v8f acc[NBLK];
#pragma unroll
  for (int g = 0; g < NBLK; ++g)
#pragma unroll
    for (int r = 0; r < 8; ++r) acc[g][r] = bias[mbase + 8 * hf + r];

  for (int kt = 0; kt < K; kt += 32) {
    if (kt + 32 < K)                            // prefetch next weight slab
      __builtin_prefetch(Wh + (size_t)(mbase + (lane & 15)) * K + kt + 32, 0, 0);
    v16h a = load_frag(Wh, K, mbase, kt, lane);
#pragma unroll
    for (int g = 0; g < NBLK; ++g) {
      v16h bf = load_frag(Xb, K, colbase + 16 * g, kt, lane);
      acc[g] = wmma_f16(a, bf, acc[g]);
    }
  }

  if (F32OUT) {
    float* out = (float*)outp;                  // [b][M][4096]
    const int col0 = colbase + (lane & 15);
#pragma unroll
    for (int g = 0; g < NBLK; ++g)
#pragma unroll
      for (int r = 0; r < 8; ++r)
        out[(size_t)(b * M + mbase + 8 * hf + r) * NPTS + col0 + 16 * g] = fmaxf(acc[g][r], 0.f);
  } else {
    h16* out = (h16*)outp;
#pragma unroll
    for (int g = 0; g < NBLK; ++g)
      store_d_relu_cm(out + (size_t)b * NPTS * M, M, colbase + 16 * g, mbase, lane, acc[g]);
  }
}

// global_feature[b][c] = max_n gf1[b][n][c]   (gf1 f16 column-major)
__global__ __launch_bounds__(256) void gmax_kernel(const h16* __restrict__ gf1,
                                                   float* __restrict__ out) {
  const int g = blockIdx.x * 256 + threadIdx.x;   // 0..4095
  const int b = g >> 10, ch = g & 1023;
  const h16* p = gf1 + (size_t)b * NPTS * 1024 + ch;
  float m = -3.0e38f;
  for (int n = 0; n < NPTS; ++n) m = fmaxf(m, (float)p[(size_t)n * 1024]);
  out[b * 1024 + ch] = m;
}

// ---------------------------------------------------------------------------
extern "C" void kernel_launch(void* const* d_in, const int* in_sizes, int n_in,
                              void* d_out, int out_size, void* d_ws, size_t ws_size,
                              hipStream_t stream) {
  const float* x   = (const float*)d_in[0];
  const float* sW0 = (const float*)d_in[1];
  const float* sb0 = (const float*)d_in[2];
  const float* sW1 = (const float*)d_in[3];
  const float* sb1 = (const float*)d_in[4];
  const float* sW2 = (const float*)d_in[5];
  const float* sb2 = (const float*)d_in[6];
  const float* gW0 = (const float*)d_in[7];
  const float* gb0 = (const float*)d_in[8];
  const float* gW1 = (const float*)d_in[9];
  const float* gb1 = (const float*)d_in[10];
  const float* mW0 = (const float*)d_in[11];
  const float* mb0 = (const float*)d_in[12];
  const float* mW1 = (const float*)d_in[13];
  const float* mb1 = (const float*)d_in[14];
  float* out = (float*)d_out;                  // [0,4096): global_feature; rest: msf

  char* ws = (char*)d_ws;                      // ~66 MB used, all fully rewritten
  int* idx    = (int*)(ws);                    //  2 MB  [B][4096][32] int
  h16* point  = (h16*)(ws + (size_t)(2)  * (1u << 20));  // 12 MB [B][4096][384]
  h16* gf0    = (h16*)(ws + (size_t)(16) * (1u << 20));  //  8 MB [B][4096][256]
  h16* gf1    = (h16*)(ws + (size_t)(24) * (1u << 20));  // 32 MB [B][4096][1024]
  h16* msf0   = (h16*)(ws + (size_t)(56) * (1u << 20));  //  8 MB [B][4096][256]
  h16* whbase = (h16*)(ws + (size_t)(64) * (1u << 20));  // ~1.4 MB f16 weights
  h16* whgW0 = whbase;                 // 256*384   = 98304
  h16* whgW1 = whbase + 98304;         // 1024*256  = 262144
  h16* whmW0 = whbase + 360448;        // 256*1024  = 262144
  h16* whmW1 = whbase + 622592;        // 128*256   = 32768
  h16* whsW1 = whbase + 655360;        // 3*64*64   = 12288
  h16* whsW2 = whbase + 667648;        // 3*128*64  = 24576

  // weight pre-conversion (independent of knn; tiny)
  cvt_f32_f16_kernel<<<(98304 + 255) / 256, 256, 0, stream>>>(gW0, whgW0, 98304);
  cvt_f32_f16_kernel<<<(262144 + 255) / 256, 256, 0, stream>>>(gW1, whgW1, 262144);
  cvt_f32_f16_kernel<<<(262144 + 255) / 256, 256, 0, stream>>>(mW0, whmW0, 262144);
  cvt_f32_f16_kernel<<<(32768 + 255) / 256, 256, 0, stream>>>(mW1, whmW1, 32768);
  cvt_f32_f16_kernel<<<(12288 + 255) / 256, 256, 0, stream>>>(sW1, whsW1, 12288);
  cvt_f32_f16_kernel<<<(24576 + 255) / 256, 256, 0, stream>>>(sW2, whsW2, 24576);

  knn_kernel<<<BATCH * (NPTS / 256), 256, 0, stream>>>(x, idx);
  msg_mlp_kernel<<<BATCH * 3 * (NPTS / 8), 256, 0, stream>>>(
      x, idx, sW0, sb0, whsW1, sb1, whsW2, sb2, point);
  // 384 -> 256
  gemm_relu<false><<<(BATCH * 16 * 64) / 8, 256, 0, stream>>>(whgW0, gb0, point, gf0, 256, 384, 16);
  // 256 -> 1024
  gemm_relu<false><<<(BATCH * 64 * 64) / 8, 256, 0, stream>>>(whgW1, gb1, gf0, gf1, 1024, 256, 64);
  // global max over N
  gmax_kernel<<<16, 256, 0, stream>>>(gf1, out);
  // 1024 -> 256
  gemm_relu<false><<<(BATCH * 16 * 64) / 8, 256, 0, stream>>>(whmW0, mb0, gf1, msf0, 256, 1024, 16);
  // 256 -> 128, f32 row-major straight into d_out
  gemm_relu<true><<<(BATCH * 8 * 64) / 8, 256, 0, stream>>>(whmW1, mb1, msf0, out + 4096, 128, 256, 8);
}
